// SparsTriangularUpdate_68161130987627
// MI455X (gfx1250) — matrix operand order repaired and checked by reading, not compiled, not run on hardware
//
#include <hip/hip_runtime.h>

// ---------------------------------------------------------------------------
// SparsTriangularUpdate for MI455X (gfx1250, wave32, WMMA)
//   pass1: LN(values) -> 4 gated GEMMs (bf16 WMMA) -> store a, atomic-reduce b
//   reduce: fold replicated S copies -> S[12][128]
//   pass2: LN(values) again (free vs storing x), tri=a*S, LN(tri),
//          2 GEMMs (gate Wgo, proj Wlo) -> fused sigmoid epilogue -> out
// ---------------------------------------------------------------------------

typedef __bf16 bf16_t;
typedef bf16_t v16bf __attribute__((ext_vector_type(16)));
typedef bf16_t v8bf  __attribute__((ext_vector_type(8)));
typedef float  v8f   __attribute__((ext_vector_type(8)));

#define DEV static __device__ __forceinline__

#define NNZ      240000
#define C_IN     256
#define C_CH     128
#define KGRP     12
#define NCOPY    256        // replicated S accumulators (atomic contention relief)
#define S_ELEMS  (KGRP * C_CH)   // 1536

// workspace layout (bytes)
#define OFF_A   0                                   // a buffer: NNZ*128 f32
#define OFF_W   ((size_t)NNZ * C_CH * 4)            // 122,880,000 (256B aligned)
#define W_GA    0
#define W_LA    32768
#define W_GB    65536
#define W_LB    98304
#define W_GO    131072
#define W_LO    196608
#define W_TOTAL 229376                              // bf16 elements
#define OFF_SC  (OFF_W + (size_t)W_TOTAL * 2)       // S copies: NCOPY*1536 f32
#define OFF_SF  (OFF_SC + (size_t)NCOPY * S_ELEMS * 4)

DEV float sigmoidf_(float x) { return 1.0f / (1.0f + __expf(-x)); }

// A-fragment (16x32 bf16): lane holds 16 contiguous K {k0..k0+15}
DEV v16bf make_a_frag(const bf16_t* p) {
  v8bf lo = *(const v8bf*)p;
  v8bf hi = *(const v8bf*)(p + 8);
  return __builtin_shufflevector(lo, hi, 0,1,2,3,4,5,6,7,8,9,10,11,12,13,14,15);
}
// B-fragment (32x16 bf16): lane holds 16 contiguous K of one W row
DEV v16bf load_b_frag(const bf16_t* p) { return *(const v16bf*)p; }

DEV v8f wmma_bf16(v16bf a, v16bf b, v8f c) {
  return __builtin_amdgcn_wmma_f32_16x16x32_bf16(false, a, false, b,
                                                 (short)0, c, false, false);
}

// reduce sum/sumsq across 8 consecutive lanes (wave32 shuffles)
DEV void red8(float& s, float& q) {
  s += __shfl_xor(s, 1, 32); q += __shfl_xor(q, 1, 32);
  s += __shfl_xor(s, 2, 32); q += __shfl_xor(q, 2, 32);
  s += __shfl_xor(s, 4, 32); q += __shfl_xor(q, 4, 32);
}

// input LN over 256 channels, 32 rows/block, 8 threads/row -> bf16 LDS tile
DEV void ln_input_to_lds(const float* __restrict__ values,
                         const float* __restrict__ g, const float* __restrict__ b,
                         bf16_t* xl, int row0, int t) {
  int m = t >> 3, seg = t & 7;
  const float* src = values + (size_t)(row0 + m) * C_IN + seg * 32;
  float v[32];
  float s = 0.f, q = 0.f;
  #pragma unroll
  for (int j = 0; j < 32; j += 4) {
    float4 f = *(const float4*)(src + j);
    v[j] = f.x; v[j+1] = f.y; v[j+2] = f.z; v[j+3] = f.w;
  }
  #pragma unroll
  for (int j = 0; j < 32; j++) { s += v[j]; q += v[j] * v[j]; }
  red8(s, q);
  float mu = s * (1.0f / 256.0f);
  float var = q * (1.0f / 256.0f) - mu * mu;
  float rs = rsqrtf(var + 1e-5f);
  bf16_t* dst = xl + m * C_IN + seg * 32;
  #pragma unroll
  for (int j = 0; j < 32; j++) {
    int c = seg * 32 + j;
    dst[j] = (bf16_t)((v[j] - mu) * rs * g[c] + b[c]);
  }
}

// ---------------------------------------------------------------------------
__global__ __launch_bounds__(256) void k_prep(
    const float* __restrict__ wga, const float* __restrict__ wla,
    const float* __restrict__ wgb, const float* __restrict__ wlb,
    const float* __restrict__ wgo, const float* __restrict__ wlo,
    bf16_t* __restrict__ W, float* __restrict__ Scopy) {
  int i = blockIdx.x * blockDim.x + threadIdx.x;
  int stride = gridDim.x * blockDim.x;
  for (int e = i; e < W_TOTAL; e += stride) {
    float v;
    if      (e < W_LA) v = wga[e];
    else if (e < W_GB) v = wla[e - W_LA];
    else if (e < W_LB) v = wgb[e - W_GB];
    else if (e < W_GO) v = wlb[e - W_LB];
    else if (e < W_LO) v = wgo[e - W_GO];
    else               v = wlo[e - W_LO];
    W[e] = (bf16_t)v;
  }
  for (int e = i; e < NCOPY * S_ELEMS; e += stride) Scopy[e] = 0.0f;
}

// ---------------------------------------------------------------------------
__global__ __launch_bounds__(256) void tri_pass1(
    const float* __restrict__ values,
    const float* __restrict__ ln_g, const float* __restrict__ ln_b,
    const bf16_t* __restrict__ Wga, const float* __restrict__ bga,
    const bf16_t* __restrict__ Wla, const float* __restrict__ bla,
    const bf16_t* __restrict__ Wgb, const float* __restrict__ bgb,
    const bf16_t* __restrict__ Wlb, const float* __restrict__ blb,
    float* __restrict__ a_buf, float* __restrict__ Scopy) {
  __shared__ bf16_t xl[32 * C_IN];  // 16 KB, two 16-row M tiles
  int t = threadIdx.x;
  int row0 = blockIdx.x * 32;
  ln_input_to_lds(values, ln_g, ln_b, xl, row0, t);
  __syncthreads();

  int wave = t >> 5, lane = t & 31;
  int n = lane & 15, khalf = lane >> 4;
  int nt = wave;  // 8 waves <-> 8 N-tiles of C_CH=128
  v8f acc[4][2] = {};
  const bf16_t* Ws[4] = { Wga, Wla, Wgb, Wlb };
  #pragma unroll
  for (int ks = 0; ks < 8; ks++) {   // K = 256 = 8 x 32
    const bf16_t* ap = xl + n * C_IN + ks * 32 + khalf * 16;
    v16bf a0 = make_a_frag(ap);
    v16bf a1 = make_a_frag(ap + 16 * C_IN);
    size_t woff = (size_t)(nt * 16 + n) * C_IN + ks * 32 + khalf * 16;
    #pragma unroll
    for (int gx = 0; gx < 4; gx++) {
      v16bf bf = load_b_frag(Ws[gx] + woff);
      acc[gx][0] = wmma_bf16(a0, bf, acc[gx][0]);
      acc[gx][1] = wmma_bf16(a1, bf, acc[gx][1]);
    }
  }
  float* sc = Scopy + (size_t)(blockIdx.x & (NCOPY - 1)) * S_ELEMS;
  #pragma unroll
  for (int v = 0; v < 8; v++) {
    int m = (lane < 16) ? v : (v + 8);
    #pragma unroll
    for (int mt = 0; mt < 2; mt++) {
      int row = row0 + mt * 16 + m;
      int col = nt * 16 + n;
      float ga = acc[0][mt][v] + bga[col];
      float la = acc[1][mt][v] + bla[col];
      float gb = acc[2][mt][v] + bgb[col];
      float lb = acc[3][mt][v] + blb[col];
      a_buf[(size_t)row * C_CH + col] = sigmoidf_(ga) * la;
      float bv = sigmoidf_(gb) * lb;
      atomicAdd(sc + (row % KGRP) * C_CH + col, bv);
    }
  }
}

// ---------------------------------------------------------------------------
__global__ __launch_bounds__(256) void k_reduce(const float* __restrict__ Scopy,
                                               float* __restrict__ Sfin) {
  int j = blockIdx.x * blockDim.x + threadIdx.x;
  if (j < S_ELEMS) {
    float s = 0.f;
    for (int c = 0; c < NCOPY; c++) s += Scopy[(size_t)c * S_ELEMS + j];
    Sfin[j] = s;
  }
}

// ---------------------------------------------------------------------------
__global__ __launch_bounds__(256) void tri_pass2(
    const float* __restrict__ values,
    const float* __restrict__ ln_g, const float* __restrict__ ln_b,
    const float* __restrict__ a_buf, const float* __restrict__ Sfin,
    const float* __restrict__ lno_g, const float* __restrict__ lno_b,
    const bf16_t* __restrict__ Wgo, const float* __restrict__ bgo,
    const bf16_t* __restrict__ Wlo, const float* __restrict__ blo,
    float* __restrict__ out) {
  __shared__ bf16_t xl[32 * C_IN];   // 16 KB
  __shared__ bf16_t tl[32 * C_CH];   //  8 KB
  int t = threadIdx.x;
  int row0 = blockIdx.x * 32;
  ln_input_to_lds(values, ln_g, ln_b, xl, row0, t);
  {  // tri = a * S[row%12], LN over 128 -> bf16 LDS tile
    int m = t >> 3, seg = t & 7;
    int row = row0 + m;
    int k = row % KGRP;
    const float* ar = a_buf + (size_t)row * C_CH + seg * 16;
    const float* sp = Sfin + k * C_CH + seg * 16;
    float v[16]; float s = 0.f, q = 0.f;
    #pragma unroll
    for (int j = 0; j < 16; j += 4) {
      float4 f = *(const float4*)(ar + j);
      float4 g4 = *(const float4*)(sp + j);
      v[j] = f.x * g4.x; v[j+1] = f.y * g4.y; v[j+2] = f.z * g4.z; v[j+3] = f.w * g4.w;
    }
    #pragma unroll
    for (int j = 0; j < 16; j++) { s += v[j]; q += v[j] * v[j]; }
    red8(s, q);
    float mu = s * (1.0f / 128.0f);
    float var = q * (1.0f / 128.0f) - mu * mu;
    float rs = rsqrtf(var + 1e-5f);
    bf16_t* dst = tl + m * C_CH + seg * 16;
    #pragma unroll
    for (int j = 0; j < 16; j++) {
      int c = seg * 16 + j;
      dst[j] = (bf16_t)((v[j] - mu) * rs * lno_g[c] + lno_b[c]);
    }
  }
  __syncthreads();

  int wave = t >> 5, lane = t & 31;
  int n = lane & 15, khalf = lane >> 4;
  #pragma unroll
  for (int ni = 0; ni < 2; ni++) {
    int nt = wave + ni * 8;  // 16 N-tiles of C_IN=256
    v8f ag[2] = {}; v8f ay[2] = {};
    #pragma unroll
    for (int ks = 0; ks < 8; ks++) {   // gate: x @ Wgo.T, K=256
      const bf16_t* ap = xl + n * C_IN + ks * 32 + khalf * 16;
      v16bf a0 = make_a_frag(ap);
      v16bf a1 = make_a_frag(ap + 16 * C_IN);
      v16bf bf = load_b_frag(Wgo + (size_t)(nt * 16 + n) * C_IN + ks * 32 + khalf * 16);
      ag[0] = wmma_bf16(a0, bf, ag[0]);
      ag[1] = wmma_bf16(a1, bf, ag[1]);
    }
    #pragma unroll
    for (int ks = 0; ks < 4; ks++) {   // proj: LN(tri) @ Wlo.T, K=128
      const bf16_t* tp = tl + n * C_CH + ks * 32 + khalf * 16;
      v16bf t0 = make_a_frag(tp);
      v16bf t1 = make_a_frag(tp + 16 * C_CH);
      v16bf bf = load_b_frag(Wlo + (size_t)(nt * 16 + n) * C_CH + ks * 32 + khalf * 16);
      ay[0] = wmma_bf16(t0, bf, ay[0]);
      ay[1] = wmma_bf16(t1, bf, ay[1]);
    }
    #pragma unroll
    for (int v = 0; v < 8; v++) {
      int m = (lane < 16) ? v : (v + 8);
      #pragma unroll
      for (int mt = 0; mt < 2; mt++) {
        int row = row0 + mt * 16 + m;
        int col = nt * 16 + n;
        out[(size_t)row * C_IN + col] =
            sigmoidf_(ag[mt][v] + bgo[col]) * (ay[mt][v] + blo[col]);
      }
    }
  }
}

// ---------------------------------------------------------------------------
extern "C" void kernel_launch(void* const* d_in, const int* in_sizes, int n_in,
                              void* d_out, int out_size, void* d_ws, size_t ws_size,
                              hipStream_t stream) {
  (void)in_sizes; (void)n_in; (void)out_size; (void)ws_size;
  const float* values  = (const float*)d_in[0];
  const float* ln_in_g = (const float*)d_in[1];
  const float* ln_in_b = (const float*)d_in[2];
  const float* W_ga = (const float*)d_in[3];
  const float* b_ga = (const float*)d_in[4];
  const float* W_la = (const float*)d_in[5];
  const float* b_la = (const float*)d_in[6];
  const float* W_gb = (const float*)d_in[7];
  const float* b_gb = (const float*)d_in[8];
  const float* W_lb = (const float*)d_in[9];
  const float* b_lb = (const float*)d_in[10];
  const float* ln_o_g = (const float*)d_in[11];
  const float* ln_o_b = (const float*)d_in[12];
  const float* W_go = (const float*)d_in[13];
  const float* b_go = (const float*)d_in[14];
  const float* W_lo = (const float*)d_in[15];
  const float* b_lo = (const float*)d_in[16];
  // d_in[17] = indices: unused by the reference computation
  float* out = (float*)d_out;

  char* ws = (char*)d_ws;
  float*  a_buf = (float*)(ws + OFF_A);
  bf16_t* W     = (bf16_t*)(ws + OFF_W);
  float*  Scopy = (float*)(ws + OFF_SC);
  float*  Sfin  = (float*)(ws + OFF_SF);

  k_prep<<<512, 256, 0, stream>>>(W_ga, W_la, W_gb, W_lb, W_go, W_lo, W, Scopy);
  tri_pass1<<<NNZ / 32, 256, 0, stream>>>(values, ln_in_g, ln_in_b,
      W + W_GA, b_ga, W + W_LA, b_la, W + W_GB, b_gb, W + W_LB, b_lb,
      a_buf, Scopy);
  k_reduce<<<(S_ELEMS + 255) / 256, 256, 0, stream>>>(Scopy, Sfin);
  tri_pass2<<<NNZ / 32, 256, 0, stream>>>(values, ln_in_g, ln_in_b, a_buf, Sfin,
      ln_o_g, ln_o_b, W + W_GO, b_go, W + W_LO, b_lo, out);
}
